// MultichannelMultiheadAttention2_82145544503594
// MI455X (gfx1250) — matrix operand order repaired and checked by reading, not compiled, not run on hardware
//
#include <hip/hip_runtime.h>
#include <hip/hip_bf16.h>

// ---------------------------------------------------------------------------
// MultichannelMultiheadAttention2 for MI455X (gfx1250, wave32, WMMA).
// All heavy contractions use v_wmma_f32_16x16x32_bf16 (bf16 in, f32 acc).
// GEMM staging uses the CDNA5 async Global->LDS path (ASYNCcnt) with LDS
// double buffering when the toolchain exposes the builtins.
// ---------------------------------------------------------------------------

#define B_     2
#define C_     16
#define AM_    4
#define HEADS_ 8
#define F_     1024
#define W_     1024
#define HD_    (F_ / HEADS_)   // 128

typedef long long i64;
typedef __attribute__((ext_vector_type(16))) __bf16 bf16x16;
typedef __attribute__((ext_vector_type(8)))  float  f32x8;
typedef int v4i_ __attribute__((vector_size(16)));   // b128 payload type

#define AS1 __attribute__((address_space(1)))
#define AS3 __attribute__((address_space(3)))

#if defined(__gfx1250__) && defined(__has_builtin)
#if __has_builtin(__builtin_amdgcn_global_load_async_to_lds_b128) && \
    __has_builtin(__builtin_amdgcn_s_wait_asynccnt)
#define HAVE_ASYNC_LDS 1
#endif
#endif
#ifndef HAVE_ASYNC_LDS
#define HAVE_ASYNC_LDS 0
#endif

// ---- fp32 -> bf16 (native conversion, RNE) --------------------------------
__device__ inline unsigned short f2bf(float x) {
  __bf16 h = (__bf16)x;
  return __builtin_bit_cast(unsigned short, h);
}

// Build a 16-element bf16 fragment from two 16-byte LDS chunks.
__device__ inline bf16x16 load_frag(const unsigned short* p0, const unsigned short* p1) {
  struct P2 { uint4 a, b; } p;
  p.a = *reinterpret_cast<const uint4*>(p0);
  p.b = *reinterpret_cast<const uint4*>(p1);
  return __builtin_bit_cast(bf16x16, p);
}

// ===========================================================================
// 3x3 conv, 16 -> 4 channels, writes y TRANSPOSED as bf16: yT[b][am][w][h]
// (K-contiguous for the downstream WMMA GEMM where K = h).
// ===========================================================================
__global__ __launch_bounds__(128) void conv3x3_c16(
    const float* __restrict__ x, const float* __restrict__ wconv,
    const float* __restrict__ bias, unsigned short* __restrict__ yT)
{
  __shared__ float xs[3][C_][130];
  __shared__ float ws[AM_][C_][9];
  __shared__ float bs[AM_];
  const int w0 = blockIdx.x * 128;
  const int h  = blockIdx.y;
  const int b  = blockIdx.z;
  const int tid = threadIdx.x;

  for (int i = tid; i < AM_ * C_ * 9; i += 128) ((float*)ws)[i] = wconv[i];
  if (tid < AM_) bs[tid] = bias[tid];
  for (int i = tid; i < 3 * C_ * 130; i += 128) {
    int r = i / (C_ * 130); int rem = i % (C_ * 130);
    int c = rem / 130;      int wl  = rem % 130;
    int hh = h - 1 + r;     int ww  = w0 - 1 + wl;
    float v = 0.f;
    if (hh >= 0 && hh < F_ && ww >= 0 && ww < W_)
      v = x[((i64)(b * C_ + c) * F_ + hh) * W_ + ww];
    xs[r][c][wl] = v;
  }
  __syncthreads();

  float acc[AM_];
  #pragma unroll
  for (int a = 0; a < AM_; ++a) acc[a] = bs[a];
  for (int c = 0; c < C_; ++c) {
    float v00 = xs[0][c][tid], v01 = xs[0][c][tid + 1], v02 = xs[0][c][tid + 2];
    float v10 = xs[1][c][tid], v11 = xs[1][c][tid + 1], v12 = xs[1][c][tid + 2];
    float v20 = xs[2][c][tid], v21 = xs[2][c][tid + 1], v22 = xs[2][c][tid + 2];
    #pragma unroll
    for (int a = 0; a < AM_; ++a) {
      const float* wk = ws[a][c];
      acc[a] += v00 * wk[0] + v01 * wk[1] + v02 * wk[2]
              + v10 * wk[3] + v11 * wk[4] + v12 * wk[5]
              + v20 * wk[6] + v21 * wk[7] + v22 * wk[8];
    }
  }
  const int w = w0 + tid;
  #pragma unroll
  for (int a = 0; a < AM_; ++a)
    yT[((i64)(b * AM_ + a) * W_ + w) * F_ + h] = f2bf(acc[a]);
}

// ===========================================================================
// Batched WMMA GEMM:  C[z](MxN) = A[z](MxK) * B'[z](NxK)^T   (f32 accumulate)
//   A: fp32 (converted to bf16 while staging) or bf16, row-major MxK
//   B': bf16, row-major NxK (mathematical B transposed, K-contiguous)
// Block tile 128x128, 8 waves (wave32), each wave 64x32 -> 8 accumulators.
// Double-buffered LDS; bf16 operands staged via GLOBAL_LOAD_ASYNC_TO_LDS.
// Optional fused epilogue: C = C*scale + prev[z] + mask  (QK^T scores).
// ===========================================================================
#define TM  128
#define TN  128
#define TKK 32
#define LD  48   // LDS row stride in halfwords (96B, 16B-aligned, bank-spread)

template <typename TA>
__global__ __launch_bounds__(256) void gemm_wmma_bf16(
    const TA* __restrict__ A, const unsigned short* __restrict__ Bt,
    float* __restrict__ C, int M, int N, int K,
    i64 aStride, int aMod, i64 bStride, i64 cStride,
    const float* __restrict__ prev, i64 prevStride,
    const float* __restrict__ mask, float scale)
{
  __shared__ alignas(16) unsigned short As[2][TM * LD];
  __shared__ alignas(16) unsigned short Bs[2][TN * LD];

  const int z = blockIdx.z;
  const TA* Ab = A + (i64)(aMod ? (z % aMod) : z) * aStride;
  const unsigned short* Bb = Bt + (i64)z * bStride;
  float* Cb = C + (i64)z * cStride;

  const int m0 = blockIdx.y * TM;
  const int n0 = blockIdx.x * TN;
  const int tid  = threadIdx.x;
  const int lane = tid & 31;
  const int wid  = tid >> 5;
  const int wm = (wid >> 2) * 64;   // wave row offset (2 wave-rows)
  const int wn = (wid & 3) * 32;    // wave col offset (4 wave-cols)

  const f32x8 zero8 = {0.f, 0.f, 0.f, 0.f, 0.f, 0.f, 0.f, 0.f};
  f32x8 acc[4][2];
  #pragma unroll
  for (int i = 0; i < 4; ++i)
    #pragma unroll
    for (int j = 0; j < 2; ++j) acc[i][j] = zero8;

  const int asel = lane >> 4;       // K-half select
  const int arow = lane & 15;       // row (A) / col (B) within 16
  const int kbA  = asel * 8;        // A frag: K = kbA+j (j<8), kbA+16+(j-8)
  const int ksB  = asel * 16;       // unused; B frag uses kbA like A

  // Stage one 128x32 A tile + 128x32 B' tile into LDS buffer `buf`.
  auto stage = [&](int buf, int k0) {
    #pragma unroll
    for (int c = tid; c < 512; c += 256) {        // 128 rows x 4 chunks of 8
      const int row = c >> 2;
      const int kc  = (c & 3) * 8;
      if constexpr (sizeof(TA) == 4) {
        // fp32 A: load + convert in VGPRs, ds_store bf16.
        const float* src = (const float*)Ab + (i64)(m0 + row) * K + (k0 + kc);
        float4 f0 = *reinterpret_cast<const float4*>(src);
        float4 f1 = *reinterpret_cast<const float4*>(src + 4);
        uint4 q;
        q.x = (unsigned)f2bf(f0.x) | ((unsigned)f2bf(f0.y) << 16);
        q.y = (unsigned)f2bf(f0.z) | ((unsigned)f2bf(f0.w) << 16);
        q.z = (unsigned)f2bf(f1.x) | ((unsigned)f2bf(f1.y) << 16);
        q.w = (unsigned)f2bf(f1.z) | ((unsigned)f2bf(f1.w) << 16);
        *reinterpret_cast<uint4*>(&As[buf][row * LD + kc]) = q;
        if (k0 + TKK < K)
          __builtin_prefetch((const char*)(src + TKK), 0, 1);
      } else {
        const unsigned short* src =
            (const unsigned short*)Ab + (i64)(m0 + row) * K + (k0 + kc);
#if HAVE_ASYNC_LDS
        __builtin_amdgcn_global_load_async_to_lds_b128(
            (AS1 v4i_*)src, (AS3 v4i_*)&As[buf][row * LD + kc], 0, 0);
#else
        *reinterpret_cast<uint4*>(&As[buf][row * LD + kc]) =
            *reinterpret_cast<const uint4*>(src);
#endif
      }
      const unsigned short* bsrc = Bb + (i64)(n0 + row) * K + (k0 + kc);
#if HAVE_ASYNC_LDS
      __builtin_amdgcn_global_load_async_to_lds_b128(
          (AS1 v4i_*)bsrc, (AS3 v4i_*)&Bs[buf][row * LD + kc], 0, 0);
#else
      *reinterpret_cast<uint4*>(&Bs[buf][row * LD + kc]) =
          *reinterpret_cast<const uint4*>(bsrc);
#endif
    }
  };

  // Async ops per wave per staged tile (in-order completion => wait <= ASY
  // guarantees the previously staged tile has fully landed in LDS).
  constexpr unsigned short ASY = (sizeof(TA) == 2) ? 4 : 2;

  const int nIter = K / TKK;
  stage(0, 0);
  for (int it = 0; it < nIter; ++it) {
    if (it + 1 < nIter) {
      stage((it + 1) & 1, (it + 1) * TKK);
#if HAVE_ASYNC_LDS
      __builtin_amdgcn_s_wait_asynccnt(ASY);
#endif
    } else {
#if HAVE_ASYNC_LDS
      __builtin_amdgcn_s_wait_asynccnt(0);
#endif
    }
    __syncthreads();                 // staged data visible to all waves

    const unsigned short* Acur = As[it & 1];
    const unsigned short* Bcur = Bs[it & 1];

    bf16x16 bfrag[2];
    #pragma unroll
    for (int nt = 0; nt < 2; ++nt) {
      const int col = wn + nt * 16 + arow;
      bfrag[nt] = load_frag(&Bcur[col * LD + kbA], &Bcur[col * LD + kbA + 16]);
    }
    #pragma unroll
    for (int mt = 0; mt < 4; ++mt) {
      const int row = wm + mt * 16 + arow;
      bf16x16 afrag = load_frag(&Acur[row * LD + kbA], &Acur[row * LD + kbA + 16]);
      acc[mt][0] = __builtin_amdgcn_wmma_f32_16x16x32_bf16(
          false, afrag, false, bfrag[0], (short)0, acc[mt][0], false, false);
      acc[mt][1] = __builtin_amdgcn_wmma_f32_16x16x32_bf16(
          false, afrag, false, bfrag[1], (short)0, acc[mt][1], false, false);
    }
    __syncthreads();                 // reads done before buffer is re-staged
  }

  // Epilogue. C layout: VGPR r holds (M = r + 8*(lane>=16), N = lane&15).
  const int csel = (lane >> 4) * 8;
  const int ccol = lane & 15;
  #pragma unroll
  for (int mt = 0; mt < 4; ++mt)
    #pragma unroll
    for (int nt = 0; nt < 2; ++nt) {
      const int col = n0 + wn + nt * 16 + ccol;
      #pragma unroll
      for (int r = 0; r < 8; ++r) {
        const int row = m0 + wm + mt * 16 + csel + r;
        const i64 idx = (i64)row * N + col;
        float v = acc[mt][nt][r];
        if (prev) v = v * scale + prev[(i64)z * prevStride + idx] + mask[idx];
        Cb[idx] = v;
      }
    }
}

// ===========================================================================
// Rotary embedding (interleaved pairs over seq axis w) + repack:
//   in : feat fp32 (b, am, f, w)
//   out: bf16 (b, am, head, w, hd)   (row-major (w, hd) per batch slice)
// ===========================================================================
__global__ __launch_bounds__(256) void rotary_pack(
    const float* __restrict__ feat, unsigned short* __restrict__ outp)
{
  const i64 i = (i64)blockIdx.x * blockDim.x + threadIdx.x;   // pair index
  const i64 total = (i64)B_ * AM_ * HEADS_ * W_ * (HD_ / 2);
  if (i >= total) return;
  int p = (int)(i % (HD_ / 2)); i64 t = i / (HD_ / 2);
  int w = (int)(t % W_); t /= W_;
  int hh = (int)(t % HEADS_); t /= HEADS_;
  int am = (int)(t % AM_); int b = (int)(t / AM_);
  const int f0 = hh * HD_ + 2 * p;
  const i64 basef = (i64)(b * AM_ + am) * F_ * W_;
  float x0 = feat[basef + (i64)f0 * W_ + w];
  float x1 = feat[basef + (i64)(f0 + 1) * W_ + w];
  float inv = __powf(10000.f, -(float)(2 * p) / (float)HD_);
  float ang = (float)w * inv;
  float s, c; __sincosf(ang, &s, &c);
  const i64 ob = (((i64)(b * AM_ + am) * HEADS_ + hh) * W_ + w) * HD_ + 2 * p;
  outp[ob]     = f2bf(x0 * c - x1 * s);
  outp[ob + 1] = f2bf(x1 * c + x0 * s);
}

// Elementwise fp32 -> bf16 copy (V path needs no layout change).
__global__ __launch_bounds__(256) void f2bf_copy(
    const float* __restrict__ in, unsigned short* __restrict__ out, i64 n)
{
  const i64 i = (i64)blockIdx.x * blockDim.x + threadIdx.x;
  if (i < n) out[i] = f2bf(in[i]);
}

// ===========================================================================
// Row softmax over qk (fp32, already scaled+biased) -> P bf16 (same layout).
// One 256-thread block (8 waves) per 1024-wide row.
// ===========================================================================
__global__ __launch_bounds__(256) void softmax_rows(
    const float* __restrict__ qk, unsigned short* __restrict__ P)
{
  __shared__ float sm[9];
  const i64 row = blockIdx.x;
  const float* src = qk + row * (i64)W_;
  unsigned short* dst = P + row * (i64)W_;
  const int tid = threadIdx.x;
  float vals[4];
  float mx = -3.4e38f;
  #pragma unroll
  for (int j = 0; j < 4; ++j) { vals[j] = src[tid + j * 256]; mx = fmaxf(mx, vals[j]); }
  #pragma unroll
  for (int off = 16; off > 0; off >>= 1) mx = fmaxf(mx, __shfl_xor(mx, off));
  if ((tid & 31) == 0) sm[tid >> 5] = mx;
  __syncthreads();
  if (tid == 0) { float m = sm[0]; for (int i = 1; i < 8; ++i) m = fmaxf(m, sm[i]); sm[8] = m; }
  __syncthreads();
  mx = sm[8];
  float s = 0.f;
  #pragma unroll
  for (int j = 0; j < 4; ++j) { vals[j] = __expf(vals[j] - mx); s += vals[j]; }
  #pragma unroll
  for (int off = 16; off > 0; off >>= 1) s += __shfl_xor(s, off);
  __syncthreads();
  if ((tid & 31) == 0) sm[tid >> 5] = s;
  __syncthreads();
  if (tid == 0) { float t = sm[0]; for (int i = 1; i < 8; ++i) t += sm[i]; sm[8] = t; }
  __syncthreads();
  const float inv = 1.f / sm[8];
  #pragma unroll
  for (int j = 0; j < 4; ++j) dst[tid + j * 256] = f2bf(vals[j] * inv);
}

// ===========================================================================
// Merge heads + depthwise channel mix (wo_dw, 4x4), write a2 TRANSPOSED:
//   aPV fp32 (b, am, head, w, hd) ;  a2T bf16 (b, o, w, f)  with f=head*HD+hd
// ===========================================================================
__global__ __launch_bounds__(256) void merge_dw(
    const float* __restrict__ aPV, const float* __restrict__ dw,
    unsigned short* __restrict__ a2T)
{
  const i64 i = (i64)blockIdx.x * blockDim.x + threadIdx.x;
  const i64 total = (i64)B_ * AM_ * W_ * F_;
  if (i >= total) return;
  int f = (int)(i % F_); i64 t = i / F_;
  int w = (int)(t % W_); t /= W_;
  int o = (int)(t % AM_); int b = (int)(t / AM_);
  const int head = f / HD_, hd = f % HD_;
  float s = 0.f;
  #pragma unroll
  for (int c = 0; c < AM_; ++c) {
    const i64 idx = (((i64)(b * AM_ + c) * HEADS_ + head) * W_ + w) * HD_ + hd;
    s += dw[o * AM_ + c] * aPV[idx];
  }
  a2T[i] = f2bf(s);
}

// ===========================================================================
// Final 3x3 conv on concat([x(16ch), a3(4ch)]) -> out (16ch).
// ===========================================================================
__global__ __launch_bounds__(128) void conv3x3_out(
    const float* __restrict__ x, const float* __restrict__ a3,
    const float* __restrict__ wconv, const float* __restrict__ bias,
    float* __restrict__ out)
{
  __shared__ float xs[3][C_ + AM_][130];
  __shared__ float ws[C_][C_ + AM_][9];
  __shared__ float bs[C_];
  const int w0 = blockIdx.x * 128;
  const int h  = blockIdx.y;
  const int b  = blockIdx.z;
  const int tid = threadIdx.x;
  const int IC = C_ + AM_;

  for (int i = tid; i < C_ * IC * 9; i += 128) ((float*)ws)[i] = wconv[i];
  if (tid < C_) bs[tid] = bias[tid];
  for (int i = tid; i < 3 * IC * 130; i += 128) {
    int r = i / (IC * 130); int rem = i % (IC * 130);
    int c = rem / 130;      int wl  = rem % 130;
    int hh = h - 1 + r;     int ww  = w0 - 1 + wl;
    float v = 0.f;
    if (hh >= 0 && hh < F_ && ww >= 0 && ww < W_) {
      if (c < C_) v = x[((i64)(b * C_ + c) * F_ + hh) * W_ + ww];
      else        v = a3[((i64)(b * AM_ + (c - C_)) * F_ + hh) * W_ + ww];
    }
    xs[r][c][wl] = v;
  }
  __syncthreads();

  float acc[C_];
  #pragma unroll
  for (int o = 0; o < C_; ++o) acc[o] = bs[o];
  for (int c = 0; c < IC; ++c) {
    float v00 = xs[0][c][tid], v01 = xs[0][c][tid + 1], v02 = xs[0][c][tid + 2];
    float v10 = xs[1][c][tid], v11 = xs[1][c][tid + 1], v12 = xs[1][c][tid + 2];
    float v20 = xs[2][c][tid], v21 = xs[2][c][tid + 1], v22 = xs[2][c][tid + 2];
    #pragma unroll
    for (int o = 0; o < C_; ++o) {
      const float* wk = ws[o][c];
      acc[o] += v00 * wk[0] + v01 * wk[1] + v02 * wk[2]
              + v10 * wk[3] + v11 * wk[4] + v12 * wk[5]
              + v20 * wk[6] + v21 * wk[7] + v22 * wk[8];
    }
  }
  const int w = w0 + tid;
  #pragma unroll
  for (int o = 0; o < C_; ++o)
    out[((i64)(b * C_ + o) * F_ + h) * W_ + w] = acc[o];
}

// ===========================================================================
// Host-side orchestration (graph-capture safe: launches only).
// ===========================================================================
extern "C" void kernel_launch(void* const* d_in, const int* in_sizes, int n_in,
                              void* d_out, int out_size, void* d_ws, size_t ws_size,
                              hipStream_t stream) {
  (void)in_sizes; (void)n_in; (void)out_size; (void)ws_size;

  const float* x       = (const float*)d_in[0];
  const float* prev_qk = (const float*)d_in[1];
  const float* mask    = (const float*)d_in[2];
  const float* wq_conv = (const float*)d_in[3];
  const float* bq_conv = (const float*)d_in[4];
  const float* wq_pw   = (const float*)d_in[5];
  const float* wk_conv = (const float*)d_in[6];
  const float* bk_conv = (const float*)d_in[7];
  const float* wk_pw   = (const float*)d_in[8];
  const float* wv_conv = (const float*)d_in[9];
  const float* bv_conv = (const float*)d_in[10];
  const float* wv_pw   = (const float*)d_in[11];
  const float* wo_pw   = (const float*)d_in[12];
  const float* wo_dw   = (const float*)d_in[13];
  const float* wo_conv = (const float*)d_in[14];
  const float* bo_conv = (const float*)d_in[15];

  float* outConv = (float*)d_out;                       // (B,C,F,W)
  float* qkOut   = outConv + (i64)B_ * C_ * F_ * W_;    // (B,AM,HEADS,W,W)

  const i64 EL = (i64)B_ * AM_ * F_ * W_;               // 8M elements
  const i64 PN = (i64)B_ * AM_ * HEADS_ * W_ * W_;      // 64M elements
  char* ws = (char*)d_ws;
  unsigned short* qbf  = (unsigned short*)(ws);                 // 16 MB
  unsigned short* kbf  = (unsigned short*)(ws + EL * 2);        // 16 MB
  unsigned short* vbf  = (unsigned short*)(ws + EL * 4);        // 16 MB
  float*          f32b = (float*)(ws + EL * 6);                 // 32 MB (feat / aPV / a3)
  unsigned short* P    = (unsigned short*)(ws + EL * 6 + EL * 4); // 128 MB
  unsigned short* yT   = P;   // conv output aliases P region (used earlier)
  unsigned short* a2T  = (unsigned short*)(ws + EL * 6 + EL * 4 + PN * 2); // 16 MB

  const dim3 cgrid(W_ / 128, F_, B_);
  const dim3 pwgrid(W_ / TN, F_ / TM, B_ * AM_);
  const i64 pairs = (i64)B_ * AM_ * HEADS_ * W_ * (HD_ / 2);

  // ---- Q path: conv -> pointwise GEMM (wmma) -> rotary ----
  conv3x3_c16<<<cgrid, 128, 0, stream>>>(x, wq_conv, bq_conv, yT);
  gemm_wmma_bf16<float><<<pwgrid, 256, 0, stream>>>(
      wq_pw, yT, f32b, F_, W_, F_,
      (i64)F_ * F_, AM_, (i64)W_ * F_, (i64)F_ * W_, nullptr, 0, nullptr, 1.f);
  rotary_pack<<<(unsigned)((pairs + 255) / 256), 256, 0, stream>>>(f32b, qbf);

  // ---- K path ----
  conv3x3_c16<<<cgrid, 128, 0, stream>>>(x, wk_conv, bk_conv, yT);
  gemm_wmma_bf16<float><<<pwgrid, 256, 0, stream>>>(
      wk_pw, yT, f32b, F_, W_, F_,
      (i64)F_ * F_, AM_, (i64)W_ * F_, (i64)F_ * W_, nullptr, 0, nullptr, 1.f);
  rotary_pack<<<(unsigned)((pairs + 255) / 256), 256, 0, stream>>>(f32b, kbf);

  // ---- V path ----
  conv3x3_c16<<<cgrid, 128, 0, stream>>>(x, wv_conv, bv_conv, yT);
  gemm_wmma_bf16<float><<<pwgrid, 256, 0, stream>>>(
      wv_pw, yT, f32b, F_, W_, F_,
      (i64)F_ * F_, AM_, (i64)W_ * F_, (i64)F_ * W_, nullptr, 0, nullptr, 1.f);
  f2bf_copy<<<(unsigned)((EL + 255) / 256), 256, 0, stream>>>(f32b, vbf, EL);

  // ---- qk = q.kT/sqrt(1024) + prev_qk + mask (fused epilogue; 2nd output) ----
  const dim3 qkgrid(W_ / TN, W_ / TM, B_ * AM_ * HEADS_);
  gemm_wmma_bf16<unsigned short><<<qkgrid, 256, 0, stream>>>(
      qbf, kbf, qkOut, W_, W_, HD_,
      (i64)W_ * HD_, 0, (i64)W_ * HD_, (i64)W_ * W_,
      prev_qk, (i64)W_ * W_, mask, 0.03125f);

  // ---- softmax rows -> P (bf16) ----
  softmax_rows<<<(unsigned)(B_ * AM_ * HEADS_ * W_), 256, 0, stream>>>(qkOut, P);

  // ---- a = P @ V  (aPV fp32 in f32b) ----
  const dim3 pvgrid(HD_ / TN, W_ / TM, B_ * AM_ * HEADS_);
  gemm_wmma_bf16<unsigned short><<<pvgrid, 256, 0, stream>>>(
      P, vbf, f32b, W_, HD_, W_,
      (i64)W_ * W_, 0, (i64)HD_ * W_, (i64)W_ * HD_, nullptr, 0, nullptr, 1.f);

  // ---- merge heads + wo_dw channel mix -> a2T (bf16, (b,o,w,f)) ----
  merge_dw<<<(unsigned)((EL + 255) / 256), 256, 0, stream>>>(f32b, wo_dw, a2T);

  // ---- wo_pw pointwise GEMM -> a3 fp32 (reuses f32b) ----
  gemm_wmma_bf16<float><<<pwgrid, 256, 0, stream>>>(
      wo_pw, a2T, f32b, F_, W_, F_,
      (i64)F_ * F_, AM_, (i64)W_ * F_, (i64)F_ * W_, nullptr, 0, nullptr, 1.f);

  // ---- final 3x3 conv on concat(x, a3) -> out ----
  conv3x3_out<<<cgrid, 128, 0, stream>>>(x, f32b, wo_conv, bo_conv, outConv);
}
